// SwitchFFNSimplified_44135083934212
// MI455X (gfx1250) — compile-verified, matrix operand order checked by reading, not run on hardware
//
#include <hip/hip_runtime.h>
#include <hip/hip_bf16.h>
#include <stdint.h>

// ---------------------------------------------------------------------------
// Problem constants (from the reference)
// ---------------------------------------------------------------------------
constexpr int Bb = 4, Ss = 2048, Dd = 1024, Ee = 8;
constexpr int Hh = 2 * Dd;           // 2048
constexpr int TT = Bb * Ss;          // 8192 tokens

// GEMM tiling
constexpr int BM = 128, BN = 128, BK = 32;
constexpr int LSTR = BK + 8;         // padded LDS row stride (bf16 elems) to dodge bank conflicts

#define USE_ASYNC_LDS 1              // CDNA5 async global->LDS staging (double buffered)

typedef __bf16 bf16_t;
typedef bf16_t v16bf __attribute__((ext_vector_type(16)));
typedef float  v8f   __attribute__((ext_vector_type(8)));

// ---------------------------------------------------------------------------
// CDNA5 async copy: 16B per lane, global -> LDS, tracked with ASYNCcnt.
// vdst = per-lane LDS byte address, vaddr = per-lane 64-bit global address.
// ---------------------------------------------------------------------------
__device__ __forceinline__ void async_b128(uint32_t lds_addr, uint64_t gaddr) {
  asm volatile("global_load_async_to_lds_b128 %0, %1, off"
               :: "v"(lds_addr), "v"(gaddr) : "memory");
}
__device__ __forceinline__ void wait_async_le4() {  // prev stage (4 instrs) retired
  asm volatile("s_wait_asynccnt 4" ::: "memory");
}
__device__ __forceinline__ void wait_async_le0() {
  asm volatile("s_wait_asynccnt 0" ::: "memory");
}
// Flat LDS address -> 32-bit LDS byte address (ISA: LDS aperture uses addr[31:0]).
__device__ __forceinline__ uint32_t lds_addr32(const void* p) {
  return (uint32_t)(uintptr_t)p;
}

// ---------------------------------------------------------------------------
// bf16 16x32 A/B fragment load from LDS, matching CDNA5 WMMA VGPR layout:
//   lanes 0-15 : row = lane,     K = {0..7} then {16..23}
//   lanes 16-31: row = lane-16,  K = {8..15} then {24..31}
// Rows are LSTR bf16 apart; each half is one 16-byte (b128) LDS read.
// ---------------------------------------------------------------------------
__device__ __forceinline__ v16bf load_frag16x32(const bf16_t* row0, int lane) {
  const int r  = lane & 15;
  const int hi = (lane >> 4) & 1;
  const bf16_t* p = row0 + r * LSTR + hi * 8;
  union { v16bf v; uint4 u[2]; } t;
  t.u[0] = *(const uint4*)(p);        // K 0-7   (or 8-15)
  t.u[1] = *(const uint4*)(p + 16);   // K 16-23 (or 24-31)
  return t.v;
}

// ---------------------------------------------------------------------------
// fp32 -> bf16 elementwise cast (vectorized: float4 in, 4x bf16 out)
// ---------------------------------------------------------------------------
__global__ void cast_f32_bf16(const float* __restrict__ src, bf16_t* __restrict__ dst, int n4) {
  int i = blockIdx.x * blockDim.x + threadIdx.x;
  if (i >= n4) return;
  float4 v = ((const float4*)src)[i];
  union { bf16_t h[4]; uint2 u; } o;
  o.h[0] = (bf16_t)v.x; o.h[1] = (bf16_t)v.y;
  o.h[2] = (bf16_t)v.z; o.h[3] = (bf16_t)v.w;
  ((uint2*)dst)[i] = o.u;
}

// ---------------------------------------------------------------------------
// Tiled transpose + cast: src[z][R][C] fp32 -> dst[z][C][R] bf16.
// Makes weight rows K-contiguous so GEMM B-staging is fully vectorized.
// ---------------------------------------------------------------------------
__global__ void transpose_cast(const float* __restrict__ src, bf16_t* __restrict__ dst,
                               int R, int C) {
  __shared__ float tile[32][33];
  const size_t mo = (size_t)blockIdx.z * R * C;
  const int c0 = blockIdx.x * 32, r0 = blockIdx.y * 32;
  const int tx = threadIdx.x, ty = threadIdx.y;      // blockDim = (32, 8)
#pragma unroll
  for (int i = 0; i < 32; i += 8)
    tile[ty + i][tx] = src[mo + (size_t)(r0 + ty + i) * C + (c0 + tx)];
  __syncthreads();
#pragma unroll
  for (int i = 0; i < 32; i += 8)
    dst[mo + (size_t)(c0 + ty + i) * R + (r0 + tx)] = (bf16_t)tile[tx][ty + i];
}

// ---------------------------------------------------------------------------
// Router: one wave per token. Coalesced x reads, 8 logit accumulators,
// cross-lane xor reduction, softmax top-1 prob, atomic expert-list build.
// ---------------------------------------------------------------------------
__global__ void router_kernel(const float* __restrict__ x, const float* __restrict__ wr,
                              const float* __restrict__ br, float* __restrict__ prob,
                              int* __restrict__ lists, int* __restrict__ counts) {
  const int lane = threadIdx.x & 31;
  const int tok = blockIdx.x * (blockDim.x >> 5) + (threadIdx.x >> 5);
  if (tok >= TT) return;
  float acc[Ee];
#pragma unroll
  for (int e = 0; e < Ee; ++e) acc[e] = 0.f;
  const float* xr = x + (size_t)tok * Dd;
  for (int j = lane; j < Dd; j += 32) {
    const float xv = xr[j];
    const float* w = wr + (size_t)j * Ee;
#pragma unroll
    for (int e = 0; e < Ee; ++e) acc[e] += xv * w[e];
  }
#pragma unroll
  for (int off = 16; off > 0; off >>= 1) {
#pragma unroll
    for (int e = 0; e < Ee; ++e) acc[e] += __shfl_xor(acc[e], off, 32);
  }
  if (lane == 0) {
    float lmax = -3.4e38f; int best = 0;
#pragma unroll
    for (int e = 0; e < Ee; ++e) {
      const float l = acc[e] + br[e];
      acc[e] = l;
      if (l > lmax) { lmax = l; best = e; }
    }
    float s = 0.f;
#pragma unroll
    for (int e = 0; e < Ee; ++e) s += __expf(acc[e] - lmax);
    prob[tok] = 1.f / s;                       // top-1 softmax prob
    const int pos = atomicAdd(&counts[best], 1);
    lists[best * TT + pos] = tok;
  }
}

// ---------------------------------------------------------------------------
// Grouped token-gather GEMM with fused epilogue.
//   FIRST:  h = relu(x_b @ w1t[e]^T + b1[e])     -> bf16 h  [TT, Hh]
//   !FIRST: y = (h  @ w2t[e]^T + b2[e]) * prob   -> fp32 out[TT, Dd]
// Block: 128x128 tile, 256 threads = 8 waves (2x4), wave tile 64x32,
// per-wave 4x2 v_wmma_f32_16x16x32_bf16 accumulators.
// Staging: double-buffered async global->LDS (4x b128 per thread per stage).
// ---------------------------------------------------------------------------
template <bool FIRST>
__global__ __launch_bounds__(256) void moe_ffn_gemm(
    const bf16_t* __restrict__ A,     // FIRST: xb [TT,Dd] ; else hb [TT,Hh]
    const bf16_t* __restrict__ Wt,    // FIRST: w1t [Ee,Hh,Dd]; else w2t [Ee,Dd,Hh]
    const float* __restrict__ bias,   // FIRST: b1 [Ee,Hh]; else b2 [Ee,Dd]
    const int* __restrict__ counts, const int* __restrict__ lists,
    const float* __restrict__ prob,
    bf16_t* __restrict__ hout, float* __restrict__ yout) {
  constexpr int K  = FIRST ? Dd : Hh;
  constexpr int Nw = FIRST ? Hh : Dd;

  const int e   = blockIdx.z;
  const int cnt = counts[e];
  const int m0  = blockIdx.y * BM;
  if (m0 >= cnt) return;                       // over-provisioned grid: idle tiles exit
  const int n0  = blockIdx.x * BN;
  const int* lst = lists + e * TT;
  const bf16_t* We = Wt + (size_t)e * Dd * Hh;

  __shared__ alignas(16) bf16_t As[2][BM * LSTR];
  __shared__ alignas(16) bf16_t Bs[2][BN * LSTR];

  const int tid  = threadIdx.x;
  const int lane = tid & 31;
  const int wid  = tid >> 5;
  const int wm   = (wid >> 2) * 64;            // wave row offset in tile
  const int wn   = (wid & 3) * 32;             // wave col offset in tile

  // Staging: each thread owns one 32-byte half-row of A and of B.
  const int srow  = tid >> 1;
  const int shalf = (tid & 1) * 16;
  const int gm    = m0 + srow;
  const int tokA  = lst[gm < cnt ? gm : cnt - 1];
  const bf16_t* aSrc = A + (size_t)tokA * K + shalf;           // gathered token row
  const bf16_t* bSrc = We + (size_t)(n0 + srow) * K + shalf;   // K-contiguous weight row

  const int sElt = srow * LSTR + shalf;
#if USE_ASYNC_LDS
  const uint32_t aL[2] = { lds_addr32(&As[0][sElt]), lds_addr32(&As[1][sElt]) };
  const uint32_t bL[2] = { lds_addr32(&Bs[0][sElt]), lds_addr32(&Bs[1][sElt]) };
  auto stage = [&](int buf, int k0) {
    const uint64_t ga = (uint64_t)(uintptr_t)(aSrc + k0);
    const uint64_t gb = (uint64_t)(uintptr_t)(bSrc + k0);
    async_b128(aL[buf],      ga);
    async_b128(aL[buf] + 16, ga + 16);
    async_b128(bL[buf],      gb);
    async_b128(bL[buf] + 16, gb + 16);
  };
#else
  bf16_t* aDst[2] = { &As[0][sElt], &As[1][sElt] };
  bf16_t* bDst[2] = { &Bs[0][sElt], &Bs[1][sElt] };
  auto stage = [&](int buf, int k0) {
    *(uint4*)(aDst[buf])     = *(const uint4*)(aSrc + k0);
    *(uint4*)(aDst[buf] + 8) = *(const uint4*)(aSrc + k0 + 8);
    *(uint4*)(bDst[buf])     = *(const uint4*)(bSrc + k0);
    *(uint4*)(bDst[buf] + 8) = *(const uint4*)(bSrc + k0 + 8);
  };
#endif

  v8f acc[4][2];
#pragma unroll
  for (int i = 0; i < 4; ++i)
#pragma unroll
    for (int j = 0; j < 2; ++j)
#pragma unroll
      for (int v = 0; v < 8; ++v) acc[i][j][v] = 0.f;

  stage(0, 0);
  int buf = 0;
  for (int k0 = 0; k0 < K; k0 += BK) {
    const bool more = (k0 + BK) < K;
    if (more) stage(buf ^ 1, k0 + BK);   // prefetch next slice into other buffer
#if USE_ASYNC_LDS
    if (more) wait_async_le4();          // 4 in flight for next stage; current stage done
    else      wait_async_le0();
#endif
    __syncthreads();                     // all waves' current stage visible in LDS

    v16bf af[4], bfr[2];
#pragma unroll
    for (int i = 0; i < 4; ++i) af[i] = load_frag16x32(&As[buf][(wm + i * 16) * LSTR], lane);
#pragma unroll
    for (int j = 0; j < 2; ++j) bfr[j] = load_frag16x32(&Bs[buf][(wn + j * 16) * LSTR], lane);
#pragma unroll
    for (int i = 0; i < 4; ++i)
#pragma unroll
      for (int j = 0; j < 2; ++j)
        acc[i][j] = __builtin_amdgcn_wmma_f32_16x16x32_bf16(
            false, af[i], false, bfr[j], (short)0, acc[i][j], false, false);
    __syncthreads();                     // safe to overwrite this buffer next round
    buf ^= 1;
  }

  // Epilogue. C/D layout: lane holds N = lane&15; VGPR v holds M = v + 8*(lane>=16).
  const int nl = lane & 15;
  const int mh = (lane >> 4) << 3;
#pragma unroll
  for (int i = 0; i < 4; ++i) {
#pragma unroll
    for (int j = 0; j < 2; ++j) {
      const int gn = n0 + wn + j * 16 + nl;
      const float bv = bias[e * Nw + gn];
#pragma unroll
      for (int v = 0; v < 8; ++v) {
        const int gmm = m0 + wm + i * 16 + mh + v;
        if (gmm < cnt) {
          const int tok = lst[gmm];
          float val = acc[i][j][v] + bv;
          if (FIRST) {
            val = fmaxf(val, 0.f);                          // ReLU
            hout[(size_t)tok * Hh + gn] = (bf16_t)val;
          } else {
            yout[(size_t)tok * Dd + gn] = val * prob[tok];  // top-1 prob scale
          }
        }
      }
    }
  }
}

// ---------------------------------------------------------------------------
// Host-side orchestration (graph-capture safe: kernels + hipMemsetAsync only)
// ---------------------------------------------------------------------------
extern "C" void kernel_launch(void* const* d_in, const int* in_sizes, int n_in,
                              void* d_out, int out_size, void* d_ws, size_t ws_size,
                              hipStream_t stream) {
  const float* x  = (const float*)d_in[0];
  const float* wr = (const float*)d_in[1];
  const float* br = (const float*)d_in[2];
  const float* w1 = (const float*)d_in[3];
  const float* b1 = (const float*)d_in[4];
  const float* w2 = (const float*)d_in[5];
  const float* b2 = (const float*)d_in[6];
  float* out = (float*)d_out;

  char* ws = (char*)d_ws;
  size_t off = 0;
  auto take = [&](size_t bytes) -> void* {
    void* p = ws + off;
    off = (off + bytes + 255) & ~(size_t)255;
    return p;
  };
  bf16_t* xb     = (bf16_t*)take((size_t)TT * Dd * sizeof(bf16_t));       // 16 MB
  bf16_t* w1t    = (bf16_t*)take((size_t)Ee * Dd * Hh * sizeof(bf16_t));  // 32 MB
  bf16_t* w2t    = (bf16_t*)take((size_t)Ee * Dd * Hh * sizeof(bf16_t));  // 32 MB
  bf16_t* hb     = (bf16_t*)take((size_t)TT * Hh * sizeof(bf16_t));       // 32 MB
  float*  prob   = (float*)take((size_t)TT * sizeof(float));
  int*    lists  = (int*)take((size_t)Ee * TT * sizeof(int));
  int*    counts = (int*)take(256);

  hipMemsetAsync(counts, 0, Ee * sizeof(int), stream);

  // 1) precision lowering + weight transposes
  {
    const int n4 = TT * Dd / 4;
    cast_f32_bf16<<<(n4 + 255) / 256, 256, 0, stream>>>(x, xb, n4);
  }
  {
    dim3 g(Hh / 32, Dd / 32, Ee), b(32, 8);
    transpose_cast<<<g, b, 0, stream>>>(w1, w1t, Dd, Hh);   // [E,D,H] -> [E,H,D]
  }
  {
    dim3 g(Dd / 32, Hh / 32, Ee), b(32, 8);
    transpose_cast<<<g, b, 0, stream>>>(w2, w2t, Hh, Dd);   // [E,H,D] -> [E,D,H]
  }

  // 2) top-1 routing + expert token lists
  router_kernel<<<TT / 8, 256, 0, stream>>>(x, wr, br, prob, lists, counts);

  // 3) grouped WMMA FFN: relu(x@w1+b1) -> h ; (h@w2+b2)*prob -> out
  {
    dim3 g(Hh / BN, TT / BM, Ee);
    moe_ffn_gemm<true><<<g, 256, 0, stream>>>(xb, w1t, b1, counts, lists, prob, hb, nullptr);
  }
  {
    dim3 g(Dd / BN, TT / BM, Ee);
    moe_ffn_gemm<false><<<g, 256, 0, stream>>>(hb, w2t, b2, counts, lists, prob, nullptr, out);
  }
}
